// MERCCell_65755949301900
// MI455X (gfx1250) — matrix operand order repaired and checked by reading, not compiled, not run on hardware
//
#include <hip/hip_runtime.h>
#include <hip/hip_bf16.h>
#include <math.h>

// ---------------------------------------------------------------- constants
#define TOK   32768      // B*S = 8*4096
#define DIM   512        // D = H = O
#define NS    4
#define SD    8

typedef _Float16 h8  __attribute__((ext_vector_type(8)));
typedef _Float16 v16h __attribute__((ext_vector_type(16)));
typedef float    v8f  __attribute__((ext_vector_type(8)));

__device__ __forceinline__ float softplusf_(float x) {
    return (x > 20.f) ? x : log1pf(expf(x));
}
__device__ __forceinline__ float gelu_exact(float x) {
    return 0.5f * x * (1.0f + erff(x * 0.70710678118654752f));
}

// A-operand 16x32 f16 per-lane fragment: halves at k = g*8+{0..7} and g*8+16+{0..7}
__device__ __forceinline__ v16h ldA16(const _Float16* __restrict__ p) {
    h8 lo = *(const h8*)p;
    h8 hi = *(const h8*)(p + 16);
    return __builtin_shufflevector(lo, hi, 0,1,2,3,4,5,6,7,8,9,10,11,12,13,14,15);
}
// B-operand 32x16 f16 per-lane fragment: halves at k = g*8+{0..7} and g*8+16+{0..7}
__device__ __forceinline__ v16h ldB16(const _Float16* __restrict__ p) {
    h8 lo = *(const h8*)p;
    h8 hi = *(const h8*)(p + 16);
    return __builtin_shufflevector(lo, hi, 0,1,2,3,4,5,6,7,8,9,10,11,12,13,14,15);
}

__device__ __forceinline__ v8f wmma_f16(v16h a, v16h b, v8f c) {
    return __builtin_amdgcn_wmma_f32_16x16x32_f16(false, a, false, b, (short)0, c, false, false);
}

// ---------------------------------------------------------------- GEMM (WMMA)
// Y[T,512] = epi( X[T,512] @ W^T )  with W row-major [512,512] in f16.
// Each wave owns a 32x64 output tile (2x4 WMMA subtiles); K=512 in steps of 32.
// MODE 0: y = gelu(acc + bias[n])            -> f16
// MODE 1: y = (acc + bias[n]) * rowscale[t]  -> f16
// MODE 2: y = acc + Yf[t,n]  (C-init from d_out small terms) -> f32 in-place
template <int MODE>
__global__ __launch_bounds__(256)
void merc_gemm512(const _Float16* __restrict__ A, const _Float16* __restrict__ Bw,
                  const float* __restrict__ bias, const float* __restrict__ rowscale,
                  _Float16* __restrict__ Yh, float* __restrict__ Yf) {
    const int lane = threadIdx.x & 31;
    const int wave = (blockIdx.x * 256 + threadIdx.x) >> 5;
    const int tn   = wave & 7;            // 512/64 = 8 column tiles
    const int tm   = wave >> 3;           // TOK/32 = 1024 row tiles
    const int lr   = lane & 15;
    const int lh   = lane >> 4;
    const int row0 = tm * 32;
    const int col0 = tn * 64;

    const _Float16* a0p = A  + (size_t)(row0 + lr) * DIM + lh * 8;
    const _Float16* a1p = a0p + (size_t)16 * DIM;
    const _Float16* bp  = Bw + (size_t)(col0 + lr) * DIM + lh * 8;

    v8f acc[2][4] = {};
    #pragma unroll 2
    for (int kt = 0; kt < DIM; kt += 32) {
        v16h a0 = ldA16(a0p + kt);
        v16h a1 = ldA16(a1p + kt);
        v16h b0 = ldB16(bp + kt);
        v16h b1 = ldB16(bp + kt + (size_t)16 * DIM);
        v16h b2 = ldB16(bp + kt + (size_t)32 * DIM);
        v16h b3 = ldB16(bp + kt + (size_t)48 * DIM);
        acc[0][0] = wmma_f16(a0, b0, acc[0][0]);
        acc[1][0] = wmma_f16(a1, b0, acc[1][0]);
        acc[0][1] = wmma_f16(a0, b1, acc[0][1]);
        acc[1][1] = wmma_f16(a1, b1, acc[1][1]);
        acc[0][2] = wmma_f16(a0, b2, acc[0][2]);
        acc[1][2] = wmma_f16(a1, b2, acc[1][2]);
        acc[0][3] = wmma_f16(a0, b3, acc[0][3]);
        acc[1][3] = wmma_f16(a1, b3, acc[1][3]);
    }

    #pragma unroll
    for (int mi = 0; mi < 2; ++mi) {
        const int trow = row0 + mi * 16 + lh * 8;
        #pragma unroll
        for (int ni = 0; ni < 4; ++ni) {
            const int o = col0 + ni * 16 + lr;
            float bo = 0.f;
            if (MODE != 2) bo = bias[o];
            #pragma unroll
            for (int j = 0; j < 8; ++j) {
                const size_t idx = (size_t)(trow + j) * DIM + o;
                float v = acc[mi][ni][j] + bo;
                if (MODE == 0) {
                    Yh[idx] = (_Float16)gelu_exact(v);
                } else if (MODE == 1) {
                    Yh[idx] = (_Float16)(v * rowscale[trow + j]);
                } else {
                    Yf[idx] = Yf[idx] + acc[mi][ni][j];
                }
            }
        }
    }
}

// ---------------------------------------------------------------- weight prep
__global__ void merc_convert_weights(const float* __restrict__ vnw1,
                                     const float* __restrict__ unw1,
                                     const float* __restrict__ vnw2,
                                     const float* __restrict__ ow,
                                     _Float16* __restrict__ w1v, _Float16* __restrict__ w1u,
                                     _Float16* __restrict__ w2v, _Float16* __restrict__ ow1) {
    const int i = blockIdx.x * blockDim.x + threadIdx.x;     // 262144 elements
    if (i < DIM * DIM) {
        w1v[i] = (_Float16)vnw1[i];
        w1u[i] = (_Float16)unw1[i];
        w2v[i] = (_Float16)vnw2[i];
        ow1[i] = (_Float16)ow[(size_t)(i >> 9) * 3072 + (i & 511)];   // out_w[:, 0:512]
    }
}

// Fold rank-1 / rank-8 feature blocks of out_w into small per-column tables.
__global__ void merc_fold_out(const float* __restrict__ ow, const float* __restrict__ outb,
                              const float* __restrict__ enw, const float* __restrict__ enb,
                              const float* __restrict__ prw, const float* __restrict__ prb,
                              const float* __restrict__ rsw, const float* __restrict__ rsb,
                              const float* __restrict__ sspw, const float* __restrict__ sspb,
                              const float* __restrict__ cspw, const float* __restrict__ cspb,
                              float* __restrict__ aen, float* __restrict__ apr,
                              float* __restrict__ ars, float* __restrict__ cvec,
                              float* __restrict__ mss, float* __restrict__ mcs) {
    const int o = blockIdx.x * blockDim.x + threadIdx.x;
    if (o >= DIM) return;
    const float* r = ow + (size_t)o * 3072;
    float en = 0.f, pr = 0.f, rs = 0.f, cv = outb[o];
    float ms[SD] = {0}, mc[SD] = {0};
    for (int h = 0; h < DIM; ++h) {
        const float w1 = r[512 + h], w2 = r[1024 + h], w3 = r[1536 + h];
        const float w4 = r[2048 + h], w5 = r[2560 + h];
        en += w1 * enw[h];
        pr += w2 * prw[h];
        rs += w3 * rsw[h];
        cv += w1 * enb[h] + w2 * prb[h] + w3 * rsb[h] + w4 * sspb[h] + w5 * cspb[h];
        #pragma unroll
        for (int j = 0; j < SD; ++j) {
            ms[j] += w4 * sspw[h * SD + j];
            mc[j] += w5 * cspw[h * SD + j];
        }
    }
    aen[o] = en; apr[o] = pr; ars[o] = rs; cvec[o] = cv;
    #pragma unroll
    for (int j = 0; j < SD; ++j) { mss[o * SD + j] = ms[j]; mcs[o * SD + j] = mc[j]; }
}

// ---------------------------------------------------------------- pointwise front
// One block per token: shared mean/var -> 4 LN branches, sb/cb banks, drive/conflict.
__global__ __launch_bounds__(128)
void merc_pointwise(const float* __restrict__ x,
                    const float* __restrict__ sbw, const float* __restrict__ sbb,
                    const float* __restrict__ sbproj, const float* __restrict__ sbprojb,
                    const float* __restrict__ sbsumw, const float* __restrict__ sbsumb,
                    const float* __restrict__ cbw, const float* __restrict__ cbb,
                    const float* __restrict__ cbproj, const float* __restrict__ cbprojb,
                    const float* __restrict__ cbsumw, const float* __restrict__ cbsumb,
                    const float* __restrict__ vnw, const float* __restrict__ vnb,
                    const float* __restrict__ unw, const float* __restrict__ unb,
                    const float* __restrict__ support_logits,
                    const float* __restrict__ conflict_weights,
                    const float* __restrict__ drive_bias,
                    _Float16* __restrict__ xv, _Float16* __restrict__ xu,
                    float* __restrict__ ss, float* __restrict__ cs,
                    float* __restrict__ drive, float* __restrict__ wconf) {
    __shared__ float sA[128], sB[128];
    __shared__ float s8[8 * 128];

    const int t = blockIdx.x;
    const int tid = threadIdx.x;
    const float* xr = x + (size_t)t * DIM;

    float xs[4], s = 0.f, q = 0.f;
    #pragma unroll
    for (int i = 0; i < 4; ++i) {
        xs[i] = xr[tid + i * 128];
        s += xs[i]; q += xs[i] * xs[i];
    }
    sA[tid] = s; sB[tid] = q;
    __syncthreads();
    for (int st = 64; st > 0; st >>= 1) {
        if (tid < st) { sA[tid] += sA[tid + st]; sB[tid] += sB[tid + st]; }
        __syncthreads();
    }
    const float mean = sA[0] * (1.f / DIM);
    const float var  = sB[0] * (1.f / DIM) - mean * mean;
    const float inv  = rsqrtf(var + 1e-5f);

    float ps[8] = {0};
    #pragma unroll
    for (int i = 0; i < 4; ++i) {
        const int d = tid + i * 128;
        const float xh = (xs[i] - mean) * inv;
        xv[(size_t)t * DIM + d] = (_Float16)(xh * vnw[d] + vnb[d]);
        xu[(size_t)t * DIM + d] = (_Float16)(xh * unw[d] + unb[d]);
        const float sv = xh * sbw[d] + sbb[d];
        const float cv = xh * cbw[d] + cbb[d];
        #pragma unroll
        for (int j = 0; j < 4; ++j) {
            ps[j]     += sv * sbproj[j * DIM + d];
            ps[4 + j] += cv * cbproj[j * DIM + d];
        }
    }
    #pragma unroll
    for (int j = 0; j < 8; ++j) s8[j * 128 + tid] = ps[j];
    __syncthreads();
    for (int st = 64; st > 0; st >>= 1) {
        if (tid < st) {
            #pragma unroll
            for (int j = 0; j < 8; ++j) s8[j * 128 + tid] += s8[j * 128 + tid + st];
        }
        __syncthreads();
    }

    if (tid == 0) {
        float sf[NS], ls[NS], cf[NS];
        #pragma unroll
        for (int j = 0; j < NS; ++j) {
            sf[j] = softplusf_(s8[j * 128] + sbprojb[j]) + 1e-4f;
            ls[j] = logf(sf[j]);
            cf[j] = softplusf_(s8[(4 + j) * 128] + cbprojb[j]);
        }
        float m1 = 0.f, m2 = 0.f, lm = 0.f, lmax = -1e30f;
        float c1 = 0.f, c2 = 0.f, cmax = -1e30f, cmin = 1e30f;
        #pragma unroll
        for (int j = 0; j < NS; ++j) {
            m1 += sf[j]; m2 += sf[j] * sf[j]; lm += ls[j];
            lmax = fmaxf(lmax, ls[j]);
            c1 += cf[j]; c2 += cf[j] * cf[j];
            cmax = fmaxf(cmax, cf[j]); cmin = fminf(cmin, cf[j]);
        }
        m1 *= 0.25f; lm *= 0.25f; c1 *= 0.25f;
        const float sstd = sqrtf(fmaxf(m2 * 0.25f - m1 * m1, 0.f));
        const float cstd = sqrtf(fmaxf(c2 * 0.25f - c1 * c1, 0.f));
        const float seedS[4] = {m1, sstd, lm, lmax};
        const float seedC[4] = {c1, cstd, cmax, cmin};
        #pragma unroll
        for (int i = 0; i < SD; ++i) {
            float a = sbsumb[i], b = cbsumb[i];
            #pragma unroll
            for (int j = 0; j < 4; ++j) {
                a += seedS[j] * sbsumw[i * 4 + j];
                b += seedC[j] * cbsumw[i * 4 + j];
            }
            ss[(size_t)t * SD + i] = a;
            cs[(size_t)t * SD + i] = b;
        }
        // softmax over support_logits, drive; softplus-weighted conflict
        float smax = support_logits[0];
        for (int j = 1; j < NS; ++j) smax = fmaxf(smax, support_logits[j]);
        float esum = 0.f, ev[NS];
        #pragma unroll
        for (int j = 0; j < NS; ++j) { ev[j] = expf(support_logits[j] - smax); esum += ev[j]; }
        float dsum = 0.f, wc = 0.f;
        #pragma unroll
        for (int j = 0; j < NS; ++j) {
            dsum += ls[j] * (ev[j] / esum);
            wc   += cf[j] * softplusf_(conflict_weights[j]);
        }
        drive[t] = drive_bias[0] + dsum * 0.5f;   // SUP_SCALE/sqrt(NS) = 1/2
        wconf[t] = wc * 0.5f;                      // 1/sqrt(NC)
    }
}

// ---------------------------------------------------------------- EML scalars + small terms
__global__ __launch_bounds__(128)
void merc_scalars(const _Float16* __restrict__ uh, const float* __restrict__ unw2,
                  const float* __restrict__ unb2,
                  const float* __restrict__ wconf, const float* __restrict__ drive,
                  const float* __restrict__ bias_res, const float* __restrict__ thresh,
                  const float* __restrict__ gam, const float* __restrict__ lam,
                  const float* __restrict__ eb,
                  const float* __restrict__ aen, const float* __restrict__ apr,
                  const float* __restrict__ ars, const float* __restrict__ cvec,
                  const float* __restrict__ mss, const float* __restrict__ mcs,
                  const float* __restrict__ ss, const float* __restrict__ cs,
                  float* __restrict__ act, float* __restrict__ out) {
    __shared__ float red[128];
    __shared__ float sc[3];
    __shared__ float s16[16];

    const int t = blockIdx.x;
    const int tid = threadIdx.x;

    float a = 0.f;
    #pragma unroll
    for (int i = 0; i < 4; ++i) {
        const int d = tid + i * 128;
        a += (float)uh[(size_t)t * DIM + d] * unw2[d];
    }
    red[tid] = a;
    __syncthreads();
    for (int st = 64; st > 0; st >>= 1) {
        if (tid < st) red[tid] += red[tid + st];
        __syncthreads();
    }
    if (tid == 0) {
        const float u    = softplusf_(red[0] + unb2[0]);
        const float res  = bias_res[0] + wconf[t] + u;
        const float eraw = gam[0] * drive[t] - lam[0] * res + eb[0];
        const float energy = 3.0f * tanhf(eraw * (1.f / 3.0f));
        const float av   = 1.f / (1.f + expf(-energy));     // TEMP = 1
        const float prec = softplusf_(energy - thresh[0]);
        act[t] = av;
        sc[0] = energy; sc[1] = prec; sc[2] = res;
    }
    if (tid < SD) {
        s16[tid]      = ss[(size_t)t * SD + tid];
        s16[SD + tid] = cs[(size_t)t * SD + tid];
    }
    __syncthreads();
    const float E = sc[0], P = sc[1], R = sc[2];
    for (int o = tid; o < DIM; o += 128) {
        float v = cvec[o] + E * aen[o] + P * apr[o] + R * ars[o];
        #pragma unroll
        for (int j = 0; j < SD; ++j)
            v += mss[o * SD + j] * s16[j] + mcs[o * SD + j] * s16[SD + j];
        out[(size_t)t * DIM + o] = v;
    }
}

// ---------------------------------------------------------------- launch
extern "C" void kernel_launch(void* const* d_in, const int* in_sizes, int n_in,
                              void* d_out, int out_size, void* d_ws, size_t ws_size,
                              hipStream_t stream) {
    const float* x        = (const float*)d_in[0];
    const float* sb_nw    = (const float*)d_in[1];
    const float* sb_nb    = (const float*)d_in[2];
    const float* sb_pw    = (const float*)d_in[3];
    const float* sb_pb    = (const float*)d_in[4];
    const float* sb_sw    = (const float*)d_in[5];
    const float* sb_sb    = (const float*)d_in[6];
    const float* cb_nw    = (const float*)d_in[7];
    const float* cb_nb    = (const float*)d_in[8];
    const float* cb_pw    = (const float*)d_in[9];
    const float* cb_pb    = (const float*)d_in[10];
    const float* cb_sw    = (const float*)d_in[11];
    const float* cb_sb    = (const float*)d_in[12];
    const float* vn_nw    = (const float*)d_in[13];
    const float* vn_nb    = (const float*)d_in[14];
    const float* vn_w1    = (const float*)d_in[15];
    const float* vn_b1    = (const float*)d_in[16];
    const float* vn_w2    = (const float*)d_in[17];
    const float* vn_b2    = (const float*)d_in[18];
    const float* un_nw    = (const float*)d_in[19];
    const float* un_nb    = (const float*)d_in[20];
    const float* un_w1    = (const float*)d_in[21];
    const float* un_b1    = (const float*)d_in[22];
    const float* un_w2    = (const float*)d_in[23];
    const float* un_b2    = (const float*)d_in[24];
    const float* en_w     = (const float*)d_in[25];
    const float* en_b     = (const float*)d_in[26];
    const float* pr_w     = (const float*)d_in[27];
    const float* pr_b     = (const float*)d_in[28];
    const float* rs_w     = (const float*)d_in[29];
    const float* rs_b     = (const float*)d_in[30];
    const float* ssp_w    = (const float*)d_in[31];
    const float* ssp_b    = (const float*)d_in[32];
    const float* csp_w    = (const float*)d_in[33];
    const float* csp_b    = (const float*)d_in[34];
    const float* out_w    = (const float*)d_in[35];
    const float* out_b    = (const float*)d_in[36];
    const float* sup_lg   = (const float*)d_in[37];
    const float* cfl_w    = (const float*)d_in[38];
    const float* drv_b    = (const float*)d_in[39];
    const float* bias_res = (const float*)d_in[40];
    const float* thresh   = (const float*)d_in[41];
    const float* gam      = (const float*)d_in[42];
    const float* lam      = (const float*)d_in[43];
    const float* ebias    = (const float*)d_in[44];

    char* ws = (char*)d_ws;
    const size_t SZ_TOK16 = (size_t)TOK * DIM * sizeof(_Float16);  // 32 MB
    _Float16* xv  = (_Float16*)(ws + 0 * SZ_TOK16);
    _Float16* xu  = (_Float16*)(ws + 1 * SZ_TOK16);  // later aliased as gv (act*value)
    _Float16* vh  = (_Float16*)(ws + 2 * SZ_TOK16);
    _Float16* uh  = (_Float16*)(ws + 3 * SZ_TOK16);
    size_t off = 4 * SZ_TOK16;
    const size_t SZ_W16 = (size_t)DIM * DIM * sizeof(_Float16);    // 512 KB
    _Float16* w1v = (_Float16*)(ws + off); off += SZ_W16;
    _Float16* w1u = (_Float16*)(ws + off); off += SZ_W16;
    _Float16* w2v = (_Float16*)(ws + off); off += SZ_W16;
    _Float16* ow1 = (_Float16*)(ws + off); off += SZ_W16;
    float* ss_b  = (float*)(ws + off); off += (size_t)TOK * SD * sizeof(float);
    float* cs_b  = (float*)(ws + off); off += (size_t)TOK * SD * sizeof(float);
    float* drive = (float*)(ws + off); off += (size_t)TOK * sizeof(float);
    float* wconf = (float*)(ws + off); off += (size_t)TOK * sizeof(float);
    float* actb  = (float*)(ws + off); off += (size_t)TOK * sizeof(float);
    float* aen   = (float*)(ws + off); off += DIM * sizeof(float);
    float* apr   = (float*)(ws + off); off += DIM * sizeof(float);
    float* ars   = (float*)(ws + off); off += DIM * sizeof(float);
    float* cvec  = (float*)(ws + off); off += DIM * sizeof(float);
    float* mss   = (float*)(ws + off); off += DIM * SD * sizeof(float);
    float* mcs   = (float*)(ws + off); off += DIM * SD * sizeof(float);

    float* out = (float*)d_out;

    // 1. weight conversion + out_w folding
    merc_convert_weights<<<(DIM * DIM + 255) / 256, 256, 0, stream>>>(
        vn_w1, un_w1, vn_w2, out_w, w1v, w1u, w2v, ow1);
    merc_fold_out<<<2, 256, 0, stream>>>(
        out_w, out_b, en_w, en_b, pr_w, pr_b, rs_w, rs_b,
        ssp_w, ssp_b, csp_w, csp_b, aen, apr, ars, cvec, mss, mcs);

    // 2. pointwise front (shared LN stats, sb/cb banks, drive/conflict)
    merc_pointwise<<<TOK, 128, 0, stream>>>(
        x, sb_nw, sb_nb, sb_pw, sb_pb, sb_sw, sb_sb,
        cb_nw, cb_nb, cb_pw, cb_pb, cb_sw, cb_sb,
        vn_nw, vn_nb, un_nw, un_nb, sup_lg, cfl_w, drv_b,
        xv, xu, ss_b, cs_b, drive, wconf);

    // 8192 waves: 1024 row tiles x 8 col tiles; 8 waves / block
    const int gemm_blocks = (TOK / 32) * (DIM / 64) / 8;   // 1024 blocks

    // 3. v_h = gelu(LNv(x) @ vn_w1^T + b1)
    merc_gemm512<0><<<gemm_blocks, 256, 0, stream>>>(xv, w1v, vn_b1, nullptr, vh, nullptr);
    // 4. u_h = gelu(LNu(x) @ un_w1^T + b1)
    merc_gemm512<0><<<gemm_blocks, 256, 0, stream>>>(xu, w1u, un_b1, nullptr, uh, nullptr);

    // 5. uncertainty + EML scalars; rank-1/rank-8 feature terms -> d_out
    merc_scalars<<<TOK, 128, 0, stream>>>(
        uh, un_w2, un_b2, wconf, drive, bias_res, thresh, gam, lam, ebias,
        aen, apr, ars, cvec, mss, mcs, ss_b, cs_b, actb, out);

    // 6. gv = activation * (v_h @ vn_w2^T + b2)   (xu buffer reused)
    merc_gemm512<1><<<gemm_blocks, 256, 0, stream>>>(vh, w2v, vn_b2, actb, xu, nullptr);

    // 7. d_out += gv @ out_w[:, :512]^T
    merc_gemm512<2><<<gemm_blocks, 256, 0, stream>>>(xu, ow1, nullptr, nullptr, nullptr, out);
}
